// SwinTransformerBlock1D_86045374808601
// MI455X (gfx1250) — compile-verified, matrix operand order checked by reading, not run on hardware
//
#include <hip/hip_runtime.h>
#include <hip/hip_bf16.h>
#include <math.h>

// ---------------------------------------------------------------------------
// Types
// ---------------------------------------------------------------------------
typedef _Float16 half_t;
typedef __attribute__((ext_vector_type(16))) _Float16 v16h;
typedef __attribute__((ext_vector_type(8)))  _Float16 v8h;
typedef __attribute__((ext_vector_type(4)))  _Float16 v4h;
typedef __attribute__((ext_vector_type(8)))  float    v8f;
typedef __attribute__((ext_vector_type(4)))  unsigned int v4u;
typedef __attribute__((ext_vector_type(8)))  unsigned int v8u;

#define C_DIM   256
#define L_DIM   16384
#define NHEAD   8
#define WIN     64
#define HDIM    32
#define SHIFT_A 32
#define HID     1024

// ---------------------------------------------------------------------------
// WMMA helpers
// ---------------------------------------------------------------------------
__device__ __forceinline__ v8f vzero8() {
  v8f z;
#pragma unroll
  for (int i = 0; i < 8; ++i) z[i] = 0.0f;
  return z;
}

__device__ __forceinline__ v8f wmma16(v16h a, v16h b, v8f c) {
  // D = A(16x32 f16) * B(32x16 f16) + C(16x16 f32)
  return __builtin_amdgcn_wmma_f32_16x16x32_f16(
      false, a, false, b, (short)0, c, false, false);
}

// Load a 16x32 f16 fragment (A layout; also used for B when the source is the
// transposed matrix stored row-major).  Per CDNA5 ISA 16-bit A layout:
//   lanes 0-15  : row = lane,    K = 0..7 then 16..23
//   lanes 16-31 : row = lane-16, K = 8..15 then 24..31
__device__ __forceinline__ v16h load_frag(const half_t* __restrict__ base, int ld) {
  const int lane = threadIdx.x & 31;
  const int row  = lane & 15;
  const int ksel = (lane >> 4) << 3;           // 0 or 8
  const half_t* p = base + row * ld + ksel;
  v8h lo = *(const v8h*)(p);
  v8h hi = *(const v8h*)(p + 16);
  v16h r;
#pragma unroll
  for (int i = 0; i < 8; ++i) { r[i] = lo[i]; r[i + 8] = hi[i]; }
  return r;
}

__device__ __forceinline__ float gelu_exact(float x) {
  return 0.5f * x * (1.0f + erff(x * 0.70710678118654752f));
}

// ---------------------------------------------------------------------------
// Weight convert + transpose:  dst[N][K] (f16) = src[K][N] (f32)
// ---------------------------------------------------------------------------
__global__ __launch_bounds__(256) void k_transpose_f16(
    const float* __restrict__ src, half_t* __restrict__ dst, int K, int N) {
  int idx = blockIdx.x * 256 + threadIdx.x;
  if (idx < K * N) {
    int k = idx / N;
    int n = idx - k * N;
    dst[n * K + k] = (half_t)src[k * N + n];
  }
}

// ---------------------------------------------------------------------------
// Fused windowed attention:  shift -> QKV -> softmax(QK^T)V -> proj -> unshift
// One workgroup (256 thr = 8 waves) per 64-token window.  Grid = B * nW = 1024.
// ---------------------------------------------------------------------------
__global__ __launch_bounds__(256) void k_attn(
    const float* __restrict__ x,       // [B, L, C]
    const half_t* __restrict__ qkvT,   // [3C, C]   (w_qkv transposed, f16)
    const half_t* __restrict__ projT,  // [C, C]    (w_proj transposed, f16)
    const float* __restrict__ b_proj,  // [C]
    float* __restrict__ out)           // [B, L, C]  (x2)
{
  __shared__ __align__(16) unsigned char smem[160 * 1024];
  half_t* Ps = (half_t*)(smem);                 // 64 KB: softmax P (phase 2+)
  half_t* Xs = (half_t*)(smem);                 // 32 KB: x window  (phase 0-1, aliases Ps)
  half_t* Qs = (half_t*)(smem + 64 * 1024);     // 32 KB: Q*scale   (phase 1-2)
  half_t* Os = (half_t*)(smem + 64 * 1024);     // 32 KB: attn out  (phase 3, aliases Qs)
  half_t* Ks = (half_t*)(smem + 96 * 1024);     // 32 KB: K
  half_t* Vt = (half_t*)(smem + 128 * 1024);    // 32 KB: V transposed [C][W]

  const int tid   = threadIdx.x;
  const int lane  = tid & 31;
  const int w     = __builtin_amdgcn_readfirstlane((int)(threadIdx.x >> 5)); // scalar wave id
  const int wid   = blockIdx.x;
  const int b     = wid >> 8;
  const int winb  = wid & 255;
  const int n0    = lane & 15;
  const int mhalf = (lane >> 4) << 3;           // 0 or 8

  // ---- Phase 0: load shifted x window -> f16 LDS -------------------------
#pragma unroll
  for (int i = 0; i < 16; ++i) {
    int f   = tid + i * 256;                    // 4096 float4 total
    int e   = f << 2;
    int row = e >> 8;
    int col = e & 255;
    int l   = (winb * WIN + row + SHIFT_A) & (L_DIM - 1);
    float4 v = *(const float4*)(x + ((size_t)b * L_DIM + l) * C_DIM + col);
    v4h h;
    h[0] = (half_t)v.x; h[1] = (half_t)v.y; h[2] = (half_t)v.z; h[3] = (half_t)v.w;
    *(v4h*)(&Xs[row * C_DIM + col]) = h;
  }
  __syncthreads();

  // ---- Phase 1: QKV GEMM  (64x256) @ (256x768) ---------------------------
  // Wave w owns output N-tiles [6w, 6w+6).  ntg<16 -> Q, <32 -> K, else V^T.
  const float scale = 0.17677669529663687f;     // 1/sqrt(32), folded into Q
#pragma unroll 1
  for (int ntl = 0; ntl < 6; ++ntl) {
    const int ntg = w * 6 + ntl;                // scalar
    if (ntl < 5) __builtin_prefetch(qkvT + (ntg + 1) * 16 * C_DIM, 0, 1);
    v16h Bf[8];
#pragma unroll
    for (int kt = 0; kt < 8; ++kt)
      Bf[kt] = load_frag(qkvT + ntg * 16 * C_DIM + kt * 32, C_DIM);
#pragma unroll 1
    for (int mt = 0; mt < 4; ++mt) {
      v8f acc = vzero8();
#pragma unroll
      for (int kt = 0; kt < 8; ++kt)
        acc = wmma16(load_frag(Xs + mt * 16 * C_DIM + kt * 32, C_DIM), Bf[kt], acc);
      const int mb = mt * 16 + mhalf;
      if (ntg < 16) {                           // scalar branch: Q (pre-scaled)
        const int n = ntg * 16 + n0;
#pragma unroll
        for (int r = 0; r < 8; ++r) Qs[(mb + r) * C_DIM + n] = (half_t)(acc[r] * scale);
      } else if (ntg < 32) {                    // K
        const int n = ntg * 16 + n0 - 256;
#pragma unroll
        for (int r = 0; r < 8; ++r) Ks[(mb + r) * C_DIM + n] = (half_t)acc[r];
      } else {                                  // V, transposed & packed
        const int n = ntg * 16 + n0 - 512;
        v8h pk;
#pragma unroll
        for (int r = 0; r < 8; ++r) pk[r] = (half_t)acc[r];
        *(v8h*)(&Vt[n * WIN + mb]) = pk;
      }
    }
  }
  __syncthreads();   // Q/K/Vt ready; Xs dead (region becomes Ps)

  // ---- Phase 2: per-head attention (wave w = head w) ---------------------
  const int h = w;
  v16h Aq[4], Bk[4];
#pragma unroll
  for (int mt = 0; mt < 4; ++mt) Aq[mt] = load_frag(Qs + mt * 16 * C_DIM + h * HDIM, C_DIM);
#pragma unroll
  for (int nt = 0; nt < 4; ++nt) Bk[nt] = load_frag(Ks + nt * 16 * C_DIM + h * HDIM, C_DIM);

  v8f S[4][4];
#pragma unroll
  for (int mt = 0; mt < 4; ++mt)
#pragma unroll
    for (int nt = 0; nt < 4; ++nt)
      S[mt][nt] = wmma16(Aq[mt], Bk[nt], vzero8());

  half_t* Pw = Ps + w * WIN * WIN;
#pragma unroll
  for (int mt = 0; mt < 4; ++mt) {
#pragma unroll
    for (int r = 0; r < 8; ++r) {
      float mx = -3.0e38f;
#pragma unroll
      for (int nt = 0; nt < 4; ++nt) mx = fmaxf(mx, S[mt][nt][r]);
#pragma unroll
      for (int off = 1; off < 16; off <<= 1)
        mx = fmaxf(mx, __shfl_xor(mx, off, 32));
      float sum = 0.0f;
#pragma unroll
      for (int nt = 0; nt < 4; ++nt) {
        float e = expf(S[mt][nt][r] - mx);
        S[mt][nt][r] = e;
        sum += e;
      }
#pragma unroll
      for (int off = 1; off < 16; off <<= 1)
        sum += __shfl_xor(sum, off, 32);
      const float inv = 1.0f / sum;
      const int m = mt * 16 + mhalf + r;
#pragma unroll
      for (int nt = 0; nt < 4; ++nt)
        Pw[m * WIN + nt * 16 + n0] = (half_t)(S[mt][nt][r] * inv);
    }
  }
  __syncthreads();   // all waves done reading Qs -> region becomes Os

  // ---- Phase 3: out_h = P (64x64) @ V_h (64x32) --------------------------
#pragma unroll
  for (int mt = 0; mt < 4; ++mt) {
#pragma unroll
    for (int nt2 = 0; nt2 < 2; ++nt2) {
      v8f acc = vzero8();
#pragma unroll
      for (int ks = 0; ks < 2; ++ks) {
        v16h Ap = load_frag(Pw + mt * 16 * WIN + ks * 32, WIN);
        v16h Bv = load_frag(Vt + (h * HDIM + nt2 * 16) * WIN + ks * 32, WIN);
        acc = wmma16(Ap, Bv, acc);
      }
      const int n = h * HDIM + nt2 * 16 + n0;
      const int mb = mt * 16 + mhalf;
#pragma unroll
      for (int r = 0; r < 8; ++r) Os[(mb + r) * C_DIM + n] = (half_t)acc[r];
    }
  }
  __syncthreads();

  // ---- Phase 4: proj GEMM + bias + reverse shift -> out ------------------
#pragma unroll 1
  for (int ntl = 0; ntl < 2; ++ntl) {
    const int ntg = w * 2 + ntl;                // scalar
    v16h Bp[8];
#pragma unroll
    for (int kt = 0; kt < 8; ++kt)
      Bp[kt] = load_frag(projT + ntg * 16 * C_DIM + kt * 32, C_DIM);
    const int n = ntg * 16 + n0;
    const float bias = b_proj[n];
#pragma unroll 1
    for (int mt = 0; mt < 4; ++mt) {
      v8f acc = vzero8();
#pragma unroll
      for (int kt = 0; kt < 8; ++kt)
        acc = wmma16(load_frag(Os + mt * 16 * C_DIM + kt * 32, C_DIM), Bp[kt], acc);
      const int mb = mt * 16 + mhalf;
#pragma unroll
      for (int r = 0; r < 8; ++r) {
        const int m = mb + r;
        const int l = (winb * WIN + m + SHIFT_A) & (L_DIM - 1);
        out[((size_t)b * L_DIM + l) * C_DIM + n] = acc[r] + bias;
      }
    }
  }
}

// ---------------------------------------------------------------------------
// LayerNorm + MLP (exact GELU) + residual, in-place over `io` (= x2 -> final).
// One workgroup per 64 rows.  Grid = B*L/64 = 1024.
// The 64x256 f32 x2 tile is DMA'd into LDS with one TDM tensor_load_to_lds
// (D# built per CDNA5 ISA), issued by wave 0.
// ---------------------------------------------------------------------------
__global__ __launch_bounds__(256) void k_mlp(
    float* __restrict__ io,             // [B*L, C]  x2 in, result out
    const half_t* __restrict__ mlp1T,   // [HID, C]  f16
    const half_t* __restrict__ mlp2T,   // [C, HID]  f16
    const float* __restrict__ gamma2,
    const float* __restrict__ beta2,
    const float* __restrict__ b_mlp1,   // [HID]
    const float* __restrict__ b_mlp2)   // [C]
{
  __shared__ __align__(16) unsigned char smem[104 * 1024];
  float*  x2s = (float*)(smem);                 // 64 KB [64][256]  (LDS offset 0)
  half_t* Lns = (half_t*)(smem + 64 * 1024);    // 32 KB [64][256]
  half_t* H1s = (half_t*)(smem + 96 * 1024);    //  8 KB [64][64]

  const int tid   = threadIdx.x;
  const int lane  = tid & 31;
  const int w     = __builtin_amdgcn_readfirstlane((int)(threadIdx.x >> 5));
  const int n0    = lane & 15;
  const int mhalf = (lane >> 4) << 3;
  const size_t row0 = (size_t)blockIdx.x * 64;

  // ---- TDM: x2 tile (64 rows x 256 f32) -> LDS ---------------------------
  if (w == 0) {
    unsigned long long ga = (unsigned long long)(const void*)(io + row0 * C_DIM);
    v4u g0;
    g0[0] = 1u;                                       // count=1, user mode
    g0[1] = 0u;                                       // lds_addr = 0 (x2s)
    g0[2] = (unsigned)(ga & 0xffffffffu);             // global_addr[31:0]
    g0[3] = (unsigned)((ga >> 32) & 0x01ffffffu) | 0x80000000u; // addr[56:32] | type=2
    v8u g1;
    g1[0] = 0x00020000u;                              // data_size = 2 (4 bytes)
    g1[1] = (unsigned)(C_DIM & 0xffff) << 16;         // tensor_dim0 = 256  (bits 79:48)
    g1[2] = (unsigned)(WIN   & 0xffff) << 16;         // tensor_dim1 = 64   (bits 111:80)
    g1[3] = (unsigned)C_DIM << 16;                    // tile_dim0 = 256    (bits 127:112)
    g1[4] = (unsigned)WIN;                            // tile_dim1 = 64
    g1[5] = (unsigned)C_DIM;                          // tensor_dim0_stride = 256
    g1[6] = 0u;
    g1[7] = 0u;
    asm volatile("tensor_load_to_lds %0, %1" : : "s"(g0), "s"(g1) : "memory");
    __builtin_amdgcn_s_wait_tensorcnt(0);
  }
  __syncthreads();

  // ---- LayerNorm: 4 threads per row --------------------------------------
  {
    const int rrow = tid >> 2;          // 0..63
    const int rq   = (tid & 3) * 64;    // quarter start column
    const float* xr = x2s + rrow * C_DIM;
    float s = 0.0f, s2 = 0.0f;
#pragma unroll 8
    for (int c = rq; c < rq + 64; ++c) { float v = xr[c]; s += v; s2 += v * v; }
    s  += __shfl_xor(s, 1, 32);  s  += __shfl_xor(s, 2, 32);
    s2 += __shfl_xor(s2, 1, 32); s2 += __shfl_xor(s2, 2, 32);
    const float mu = s * (1.0f / C_DIM);
    const float rs = rsqrtf(s2 * (1.0f / C_DIM) - mu * mu + 1e-5f);
#pragma unroll 8
    for (int c = rq; c < rq + 64; ++c)
      Lns[rrow * C_DIM + c] = (half_t)((xr[c] - mu) * rs * gamma2[c] + beta2[c]);
  }
  __syncthreads();

  // ---- MLP: hidden streamed in 64-wide chunks ----------------------------
  v8f acc[8];
#pragma unroll
  for (int i = 0; i < 8; ++i) acc[i] = vzero8();

#pragma unroll 1
  for (int ch = 0; ch < 16; ++ch) {
    // GEMM1 + GELU -> H1s.  16 tiles (4M x 4N), wave w does tiles 2w, 2w+1.
#pragma unroll 1
    for (int tt = 0; tt < 2; ++tt) {
      const int t  = w * 2 + tt;                // scalar
      const int mt = t >> 2;
      const int nt = t & 3;
      if (ch < 15)
        __builtin_prefetch(mlp1T + ((ch + 1) * 64 + nt * 16) * C_DIM, 0, 1);
      v8f a1 = vzero8();
#pragma unroll
      for (int kt = 0; kt < 8; ++kt)
        a1 = wmma16(load_frag(Lns + mt * 16 * C_DIM + kt * 32, C_DIM),
                    load_frag(mlp1T + (ch * 64 + nt * 16) * C_DIM + kt * 32, C_DIM),
                    a1);
      const int nloc = nt * 16 + n0;
      const float b1 = b_mlp1[ch * 64 + nloc];
      const int mb = mt * 16 + mhalf;
#pragma unroll
      for (int r = 0; r < 8; ++r)
        H1s[(mb + r) * 64 + nloc] = (half_t)gelu_exact(a1[r] + b1);
    }
    __syncthreads();

    // GEMM2 accumulate: out(64x256) += H1(64x64) @ mlp2[ch,:]
#pragma unroll 1
    for (int ntl = 0; ntl < 2; ++ntl) {
      const int ntg = w * 2 + ntl;              // scalar
      if (ch < 15)
        __builtin_prefetch(mlp2T + ntg * 16 * HID + (ch + 1) * 64, 0, 1);
      v16h B0 = load_frag(mlp2T + ntg * 16 * HID + ch * 64,      HID);
      v16h B1 = load_frag(mlp2T + ntg * 16 * HID + ch * 64 + 32, HID);
#pragma unroll
      for (int mt = 0; mt < 4; ++mt) {
        v16h A0 = load_frag(H1s + mt * 16 * 64,      64);
        v16h A1 = load_frag(H1s + mt * 16 * 64 + 32, 64);
        const int ai = mt * 2 + ntl;
        acc[ai] = wmma16(A0, B0, acc[ai]);
        acc[ai] = wmma16(A1, B1, acc[ai]);
      }
    }
    __syncthreads();
  }

  // ---- epilogue: + b_mlp2 + residual -------------------------------------
#pragma unroll
  for (int ai = 0; ai < 8; ++ai) {
    const int mt  = ai >> 1;
    const int ntg = w * 2 + (ai & 1);
    const int n   = ntg * 16 + n0;
    const float b2 = b_mlp2[n];
    const int mb = mt * 16 + mhalf;
#pragma unroll
    for (int r = 0; r < 8; ++r) {
      const int m = mb + r;
      io[(row0 + m) * C_DIM + n] = acc[ai][r] + b2 + x2s[m * C_DIM + n];
    }
  }
}

// ---------------------------------------------------------------------------
// Host launcher
// ---------------------------------------------------------------------------
extern "C" void kernel_launch(void* const* d_in, const int* in_sizes, int n_in,
                              void* d_out, int out_size, void* d_ws, size_t ws_size,
                              hipStream_t stream) {
  const float* x      = (const float*)d_in[0];
  const float* w_qkv  = (const float*)d_in[1];
  const float* w_proj = (const float*)d_in[2];
  const float* b_proj = (const float*)d_in[3];
  const float* gamma2 = (const float*)d_in[4];
  const float* beta2  = (const float*)d_in[5];
  const float* w_mlp1 = (const float*)d_in[6];
  const float* b_mlp1 = (const float*)d_in[7];
  const float* w_mlp2 = (const float*)d_in[8];
  const float* b_mlp2 = (const float*)d_in[9];

  half_t* ws    = (half_t*)d_ws;
  half_t* qkvT  = ws;                 // [768][256]  196608
  half_t* projT = ws + 196608;        // [256][256]   65536
  half_t* m1T   = ws + 262144;        // [1024][256] 262144
  half_t* m2T   = ws + 524288;        // [256][1024] 262144

  k_transpose_f16<<<768,  256, 0, stream>>>(w_qkv,  qkvT, 256,  768);
  k_transpose_f16<<<256,  256, 0, stream>>>(w_proj, projT, 256, 256);
  k_transpose_f16<<<1024, 256, 0, stream>>>(w_mlp1, m1T,  256, 1024);
  k_transpose_f16<<<1024, 256, 0, stream>>>(w_mlp2, m2T, 1024,  256);

  k_attn<<<1024, 256, 0, stream>>>(x, qkvT, projT, b_proj, (float*)d_out);
  k_mlp <<<1024, 256, 0, stream>>>((float*)d_out, m1T, m2T,
                                   gamma2, beta2, b_mlp1, b_mlp2);
}
